// SuperpixelNeighborhoodAttention_27307402068264
// MI455X (gfx1250) — compile-verified, matrix operand order checked by reading, not run on hardware
//
#include <hip/hip_runtime.h>
#include <hip/hip_bf16.h>

// ---------------------------------------------------------------------------
// Superpixel Neighborhood Attention for MI455X (gfx1250, wave32, WMMA).
//   B=8, C=128, H=W=128, 4 heads x headdim 32, 7x7 NATTEN window, 16x16 SP grid
// All contractions use v_wmma_f32_16x16x32_bf16 (f32 accumulate).
// GEMM tile staging uses GLOBAL_LOAD_ASYNC_TO_LDS_B128 (ASYNCcnt path).
// ---------------------------------------------------------------------------

#define Bc   8
#define Cc   128
#define Hc   128
#define Wc   128
#define NPIX (Bc * Hc * Wc)          // 131072
#define NH   4
#define HD   32
#define KW   7
#define QSCALE 0.17677669529663687f  // 32^-0.5

typedef __attribute__((ext_vector_type(16))) __bf16 v16bf;
typedef __attribute__((ext_vector_type(8)))  __bf16 v8bf;
typedef __attribute__((ext_vector_type(8)))  float  v8f;

__device__ __forceinline__ v16bf mk16(v8bf lo, v8bf hi) {
  v16bf r;
#pragma unroll
  for (int i = 0; i < 8; ++i) { r[i] = lo[i]; r[i + 8] = hi[i]; }
  return r;
}

__device__ __forceinline__ v8f wmma_bf16(v16bf a, v16bf b, v8f c) {
  return __builtin_amdgcn_wmma_f32_16x16x32_bf16(false, a, false, b,
                                                 (short)0, c, false, false);
}

__device__ __forceinline__ int iclamp(int v, int lo, int hi) {
  return v < lo ? lo : (v > hi ? hi : v);
}

// Async 16B global -> LDS copy (per-lane addresses, GV mode).
// LDS operand is the wave-relative LDS byte address = low 32 bits of the
// generic pointer (ISA 10.2: LDS aperture uses addr[31:0]).
__device__ __forceinline__ void async_copy_b128(const void* gptr, void* lptr) {
  unsigned lds = (unsigned)(uintptr_t)lptr;
  asm volatile("global_load_async_to_lds_b128 %0, %1, off"
               :: "v"(lds), "v"(gptr) : "memory");
}

__device__ __forceinline__ void wait_async0() {
#if __has_builtin(__builtin_amdgcn_s_wait_asynccnt)
  __builtin_amdgcn_s_wait_asynccnt(0);
#else
  asm volatile("s_wait_asynccnt 0" ::: "memory");
#endif
}

// ---------------------------------------------------------------------------
// K0: weights fp32 -> bf16.  wqkv = [qk_w(256x128); v_w(128x128)] rows=out_ch.
// ---------------------------------------------------------------------------
__global__ void wconv_kernel(const float* __restrict__ qk,
                             const float* __restrict__ vw,
                             const float* __restrict__ pw,
                             __bf16* __restrict__ wqkv,
                             __bf16* __restrict__ wproj) {
  int i = blockIdx.x * 256 + threadIdx.x;           // 65536 total
  if (i < 49152) {
    wqkv[i] = (__bf16)((i < 32768) ? qk[i] : vw[i - 32768]);
  } else {
    int j = i - 49152;
    wproj[j] = (__bf16)pw[j];
  }
}

// ---------------------------------------------------------------------------
// K1: x[B,C,H,W] fp32 -> xbf[pix, C] bf16 (LDS-tiled transpose, padded banks).
// One block per (b,h): transposes the 128(c) x 128(w) slice.
// ---------------------------------------------------------------------------
__global__ __launch_bounds__(256) void xpose_kernel(const float* __restrict__ x,
                                                    __bf16* __restrict__ xbf) {
  __shared__ __bf16 T[128 * 130];
  int bh = blockIdx.x;
  int b = bh >> 7, h = bh & 127;
  size_t xbase = (size_t)b * (128 * 128 * 128) + (size_t)h * 128;
  for (int i = threadIdx.x; i < 16384; i += 256) {
    int c = i >> 7, w = i & 127;
    T[c * 130 + w] = (__bf16)x[xbase + (size_t)c * 16384 + w];
  }
  __syncthreads();
  size_t pbase = ((size_t)(b * 128 + h)) * 128;     // pixel base (w=0)
  for (int i = threadIdx.x; i < 16384; i += 256) {
    int w = i >> 7, c = i & 127;
    xbf[(pbase + w) * 128 + c] = T[c * 130 + w];
  }
}

// ---------------------------------------------------------------------------
// K2: per-pixel argmax over 256 superpixel sims -> L[pix].
// One wave32 per pixel: 8 values/lane, butterfly reduce (first-max tiebreak).
// ---------------------------------------------------------------------------
__global__ __launch_bounds__(256) void spx_argmax_kernel(
    const float* __restrict__ sims, int* __restrict__ L) {
  int wv = threadIdx.x >> 5, lane = threadIdx.x & 31;
  int pix = blockIdx.x * 8 + wv;
  const float* s = sims + (size_t)pix * 256 + lane * 8;
  float bv = -1.0f; int bi = lane * 8;
#pragma unroll
  for (int i = 0; i < 8; ++i) {
    float v = s[i];
    if (v > bv) { bv = v; bi = lane * 8 + i; }
  }
#pragma unroll
  for (int d = 1; d < 32; d <<= 1) {
    float ov = __shfl_xor(bv, d, 32);
    int   oi = __shfl_xor(bi, d, 32);
    if (ov > bv || (ov == bv && oi < bi)) { bv = ov; bi = oi; }
  }
  if (lane == 0) L[pix] = bi;
}

// ---------------------------------------------------------------------------
// K3: fused QKV projection GEMM: qkv[pix, 0:384] = xbf[pix,:] @ wqkv^T
//     (q cols 0..127 pre-scaled by 32^-0.5).
// Block tile 128x128, 8 waves, each wave 32x64 = 2x4 wmma tiles.
// Tiles staged via async global->LDS B128 copies (ASYNCcnt).
// ---------------------------------------------------------------------------
__global__ __launch_bounds__(256) void qkv_gemm_kernel(
    const __bf16* __restrict__ A, const __bf16* __restrict__ Wt,
    __bf16* __restrict__ qkv) {
  __shared__ __bf16 As[16384];
  __shared__ __bf16 Bs[16384];
  int m0 = blockIdx.x * 128, n0 = blockIdx.y * 128;
  int tid = threadIdx.x;
  {
    const uint4* ga = (const uint4*)(A + (size_t)m0 * 128);
    const uint4* gb = (const uint4*)(Wt + (size_t)n0 * 128);
    uint4* sa = (uint4*)As; uint4* sb = (uint4*)Bs;
#pragma unroll
    for (int i = 0; i < 8; ++i) {
      async_copy_b128(ga + tid + i * 256, sa + tid + i * 256);
      async_copy_b128(gb + tid + i * 256, sb + tid + i * 256);
    }
  }
  wait_async0();
  __syncthreads();

  int wave = tid >> 5, lane = tid & 31;
  int l16 = lane & 15, koff = (lane & 16) ? 8 : 0, mhi = (lane & 16) ? 8 : 0;
  int ms = (wave & 3) * 32, ns = (wave >> 2) * 64;

  v8f acc[2][4];
#pragma unroll
  for (int mi = 0; mi < 2; ++mi)
#pragma unroll
    for (int ni = 0; ni < 4; ++ni) acc[mi][ni] = (v8f){0,0,0,0,0,0,0,0};

#pragma unroll
  for (int kb = 0; kb < 4; ++kb) {
    int k0 = kb * 32 + koff;
    v16bf af[2], bfr[4];
#pragma unroll
    for (int mi = 0; mi < 2; ++mi) {
      const __bf16* p = As + (ms + mi * 16 + l16) * 128 + k0;
      af[mi] = mk16(*(const v8bf*)p, *(const v8bf*)(p + 16));
    }
#pragma unroll
    for (int ni = 0; ni < 4; ++ni) {
      const __bf16* p = Bs + (ns + ni * 16 + l16) * 128 + k0;
      bfr[ni] = mk16(*(const v8bf*)p, *(const v8bf*)(p + 16));
    }
#pragma unroll
    for (int mi = 0; mi < 2; ++mi)
#pragma unroll
      for (int ni = 0; ni < 4; ++ni)
        acc[mi][ni] = wmma_bf16(af[mi], bfr[ni], acc[mi][ni]);
  }

  float scale = (n0 == 0) ? QSCALE : 1.0f;   // block-uniform: q cols are block y==0
#pragma unroll
  for (int mi = 0; mi < 2; ++mi)
#pragma unroll
    for (int ni = 0; ni < 4; ++ni)
#pragma unroll
      for (int j = 0; j < 8; ++j) {
        int pix = m0 + ms + mi * 16 + j + mhi;
        int nc  = n0 + ns + ni * 16 + l16;
        qkv[(size_t)pix * 384 + nc] = (__bf16)(acc[mi][ni][j] * scale);
      }
}

// ---------------------------------------------------------------------------
// K4: NATTEN 7x7 + superpixel-masked softmax, full-WMMA attention.
// Block = (w-tile of 16 queries, h, b), 128 threads = 4 waves = 4 heads.
// Key union of tile = 7 rows x 22 cols = 154 positions, padded to 160.
// S = Q(16x32) x K^T : 10 wmma.  O^T = V^T(32x32k) x P^T : 10 wmma.
// ---------------------------------------------------------------------------
__global__ __launch_bounds__(128) void attn_kernel(
    const __bf16* __restrict__ qkv, const int* __restrict__ L,
    __bf16* __restrict__ obf) {
  __shared__ __bf16 Vt[128 * 160];        // [channel][key]  (40 KB)
  __shared__ __bf16 pS[4 * 16 * 160];     // [head][query][key] (20 KB)

  int b = blockIdx.z, h = blockIdx.y, w0 = blockIdx.x * 16;
  int tid = threadIdx.x;
  int head = tid >> 5, lane = tid & 31;
  int l16 = lane & 15, koff = (lane & 16) ? 8 : 0, mhi = (lane & 16) ? 8 : 0;

  int colbase = iclamp(w0 - 3, 0, Wc - KW);
  int hh0     = iclamp(h - 3, 0, Hc - KW);

  // ---- stage V (all heads) into LDS transposed: Vt[ch][key] ----------------
  // While staging, prefetch the K-part rows the S-GEMM will gather next
  // (global_prefetch_b8 path).
#pragma unroll
  for (int p = 0; p < 10; ++p) {
    int n   = p * 16 + (tid >> 3);
    int ncl = n < 154 ? n : 153;
    int r   = ncl / 22, ci = ncl - r * 22;
    int c   = iclamp(colbase + ci, 0, Wc - 1);
    int kpix = (b * 128 + hh0 + r) * 128 + c;
    int ch0 = (tid & 7) * 16;
    const __bf16* src = qkv + (size_t)kpix * 384 + 256 + ch0;
    __builtin_prefetch(qkv + (size_t)kpix * 384 + 128 + ch0, 0, 3);
    v8bf lo = *(const v8bf*)src;
    v8bf hi = *(const v8bf*)(src + 8);
#pragma unroll
    for (int i = 0; i < 8; ++i) {
      Vt[(ch0 + i) * 160 + n]     = lo[i];
      Vt[(ch0 + 8 + i) * 160 + n] = hi[i];
    }
  }
  __syncthreads();

  // ---- Q fragment (A-matrix layout, K = head dim = 32) ---------------------
  int wq0  = w0 + l16;
  int pixq = (b * 128 + h) * 128 + wq0;
  const __bf16* qb = qkv + (size_t)pixq * 384 + head * 32 + koff;
  v16bf qf = mk16(*(const v8bf*)qb, *(const v8bf*)(qb + 16));

  // ---- S = Q x K^T over 10 key n-tiles ------------------------------------
  v8f acc[10];
  int cA[10], LnA[10];
  bool vA[10];
#pragma unroll
  for (int t = 0; t < 10; ++t) {
    int nn  = t * 16 + l16;
    bool vn = nn < 154;
    int ncl = vn ? nn : 153;
    int r   = ncl / 22, ci = ncl - r * 22;
    int c   = colbase + ci;
    int ccl = c < Wc - 1 ? c : Wc - 1;
    int kpix = (b * 128 + hh0 + r) * 128 + ccl;
    cA[t] = c; vA[t] = vn; LnA[t] = L[kpix];
    const __bf16* kb = qkv + (size_t)kpix * 384 + 128 + head * 32 + koff;
    v16bf kf = mk16(*(const v8bf*)kb, *(const v8bf*)(kb + 16));
    v8f z = (v8f){0,0,0,0,0,0,0,0};
    acc[t] = wmma_bf16(qf, kf, z);
  }

  // ---- masked softmax per query row (C layout: lane = key, elem j = query) -
#pragma unroll
  for (int j = 0; j < 8; ++j) {
    int mq  = j + mhi;
    int wq  = w0 + mq;
    int sws = iclamp(wq - 3, 0, Wc - KW);
    int slab = (h >> 3) * 16 + (wq >> 3);
    float lv[10];
    float mx = -3.0e38f;
#pragma unroll
    for (int t = 0; t < 10; ++t) {
      bool ok = vA[t] && (cA[t] >= sws) && (cA[t] < sws + KW);
      lv[t] = ok ? acc[t][j] : -3.0e38f;
      mx = fmaxf(mx, lv[t]);
    }
#pragma unroll
    for (int d = 1; d < 16; d <<= 1) mx = fmaxf(mx, __shfl_xor(mx, d, 32));
    float e[10];
    float s1 = 0.f, s2 = 0.f;
#pragma unroll
    for (int t = 0; t < 10; ++t) {
      float ev = (lv[t] > -1.0e37f) ? __expf(lv[t] - mx) : 0.f;
      e[t] = ev;
      s1 += ev;
      s2 += (LnA[t] == slab) ? ev : 0.f;
    }
#pragma unroll
    for (int d = 1; d < 16; d <<= 1) {
      s1 += __shfl_xor(s1, d, 32);
      s2 += __shfl_xor(s2, d, 32);
    }
    // softmax * pi, renormalized: p = e*pi / (s2 + 1e-8*s1)
    float inv = 1.0f / (s2 + 1.0e-8f * s1);
#pragma unroll
    for (int t = 0; t < 10; ++t) {
      float pv = (LnA[t] == slab) ? e[t] * inv : 0.f;
      pS[(head * 16 + mq) * 160 + t * 16 + l16] = (__bf16)pv;
    }
  }
  __syncthreads();

  // ---- O^T = V^T x P^T  (A = V^T from Vt, B = P^T from pS, both contiguous)-
#pragma unroll
  for (int dt = 0; dt < 2; ++dt) {
    v8f oc = (v8f){0,0,0,0,0,0,0,0};
#pragma unroll
    for (int kc = 0; kc < 5; ++kc) {
      int kk = kc * 32 + koff;
      const __bf16* va = Vt + (head * 32 + dt * 16 + l16) * 160 + kk;
      v16bf af = mk16(*(const v8bf*)va, *(const v8bf*)(va + 16));
      const __bf16* pb = pS + (head * 16 + l16) * 160 + kk;
      v16bf bfm = mk16(*(const v8bf*)pb, *(const v8bf*)(pb + 16));
      oc = wmma_bf16(af, bfm, oc);
    }
    // C layout: lane = query (N), elem j = dim (M)
#pragma unroll
    for (int j = 0; j < 8; ++j) {
      int dim = dt * 16 + j + mhi;
      obf[(size_t)pixq * 128 + head * 32 + dim] = (__bf16)oc[j];
    }
  }
}

// ---------------------------------------------------------------------------
// K5: output projection GEMM + NHWC->NCHW scatter (fp32 out).
// ---------------------------------------------------------------------------
__global__ __launch_bounds__(256) void proj_gemm_kernel(
    const __bf16* __restrict__ A, const __bf16* __restrict__ Wt,
    float* __restrict__ out) {
  __shared__ __bf16 As[16384];
  __shared__ __bf16 Bs[16384];
  int m0 = blockIdx.x * 128;
  int tid = threadIdx.x;
  {
    const uint4* ga = (const uint4*)(A + (size_t)m0 * 128);
    const uint4* gb = (const uint4*)Wt;
    uint4* sa = (uint4*)As; uint4* sb = (uint4*)Bs;
#pragma unroll
    for (int i = 0; i < 8; ++i) {
      async_copy_b128(ga + tid + i * 256, sa + tid + i * 256);
      async_copy_b128(gb + tid + i * 256, sb + tid + i * 256);
    }
  }
  wait_async0();
  __syncthreads();

  int wave = tid >> 5, lane = tid & 31;
  int l16 = lane & 15, koff = (lane & 16) ? 8 : 0, mhi = (lane & 16) ? 8 : 0;
  int ms = (wave & 3) * 32, ns = (wave >> 2) * 64;

  v8f acc[2][4];
#pragma unroll
  for (int mi = 0; mi < 2; ++mi)
#pragma unroll
    for (int ni = 0; ni < 4; ++ni) acc[mi][ni] = (v8f){0,0,0,0,0,0,0,0};

#pragma unroll
  for (int kb = 0; kb < 4; ++kb) {
    int k0 = kb * 32 + koff;
    v16bf af[2], bfr[4];
#pragma unroll
    for (int mi = 0; mi < 2; ++mi) {
      const __bf16* p = As + (ms + mi * 16 + l16) * 128 + k0;
      af[mi] = mk16(*(const v8bf*)p, *(const v8bf*)(p + 16));
    }
#pragma unroll
    for (int ni = 0; ni < 4; ++ni) {
      const __bf16* p = Bs + (ns + ni * 16 + l16) * 128 + k0;
      bfr[ni] = mk16(*(const v8bf*)p, *(const v8bf*)(p + 16));
    }
#pragma unroll
    for (int mi = 0; mi < 2; ++mi)
#pragma unroll
      for (int ni = 0; ni < 4; ++ni)
        acc[mi][ni] = wmma_bf16(af[mi], bfr[ni], acc[mi][ni]);
  }

#pragma unroll
  for (int mi = 0; mi < 2; ++mi)
#pragma unroll
    for (int ni = 0; ni < 4; ++ni)
#pragma unroll
      for (int j = 0; j < 8; ++j) {
        int pix = m0 + ms + mi * 16 + j + mhi;
        int c   = ns + ni * 16 + l16;
        int w = pix & 127, hh = (pix >> 7) & 127, bb = pix >> 14;
        out[(((size_t)(bb * 128 + c)) * 128 + hh) * 128 + w] = acc[mi][ni][j];
      }
}

// ---------------------------------------------------------------------------
// Host launcher. Inputs: x, sims, qk_w, v_w, proj_w (all fp32).
// Workspace: xbf 32MB | wqkv 96KB | wproj 32KB | qkv 96MB | L 512KB | obf 32MB
// ---------------------------------------------------------------------------
extern "C" void kernel_launch(void* const* d_in, const int* in_sizes, int n_in,
                              void* d_out, int out_size, void* d_ws, size_t ws_size,
                              hipStream_t stream) {
  (void)in_sizes; (void)n_in; (void)out_size; (void)ws_size;
  const float* x     = (const float*)d_in[0];
  const float* sims  = (const float*)d_in[1];
  const float* qk_w  = (const float*)d_in[2];
  const float* v_w   = (const float*)d_in[3];
  const float* pr_w  = (const float*)d_in[4];
  float* out = (float*)d_out;

  char* ws = (char*)d_ws;
  size_t off = 0;
  auto carve = [&](size_t bytes) {
    char* p = ws + off;
    off += (bytes + 255) & ~(size_t)255;
    return p;
  };
  __bf16* xbf   = (__bf16*)carve((size_t)NPIX * 128 * 2);
  __bf16* wqkv  = (__bf16*)carve((size_t)384 * 128 * 2);
  __bf16* wproj = (__bf16*)carve((size_t)128 * 128 * 2);
  __bf16* qkv   = (__bf16*)carve((size_t)NPIX * 384 * 2);
  int*    Lbuf  = (int*)   carve((size_t)NPIX * 4);
  __bf16* obf   = (__bf16*)carve((size_t)NPIX * 128 * 2);

  wconv_kernel<<<256, 256, 0, stream>>>(qk_w, v_w, pr_w, wqkv, wproj);
  xpose_kernel<<<Bc * Hc, 256, 0, stream>>>(x, xbf);
  spx_argmax_kernel<<<NPIX / 8, 256, 0, stream>>>(sims, Lbuf);
  qkv_gemm_kernel<<<dim3(NPIX / 128, 3), 256, 0, stream>>>(xbf, wqkv, qkv);
  attn_kernel<<<dim3(Wc / 16, Hc, Bc), 128, 0, stream>>>(qkv, Lbuf, obf);
  proj_gemm_kernel<<<dim3(NPIX / 128, 1), 256, 0, stream>>>(obf, wproj, out);
}
